// MPNN3D_28681791603323
// MI455X (gfx1250) — matrix operand order repaired and checked by reading, not compiled, not run on hardware
//
#include <hip/hip_runtime.h>

// ---------------------------------------------------------------------------
// MPNN on gfx1250 (MI455X). All GEMMs use v_wmma_f32_16x16x32_f16; per-layer
// weight fragments are staged into LDS via the Tensor Data Mover
// (tensor_load_to_lds + s_wait_tensorcnt), B-fragments then read with ds_load.
//
// Decomposition (exact):  mi @ msg_W1 = (f@W1a)[src] + (f@W1b)[dst] + w@W1c + sq*w1d
// so no per-edge gather of f16 features is needed; only f32 P/Q row gathers.
//
// Fragment layout assumptions (CDNA5 ISA 7.12.2, wave32):
//   A (16x32 f16): lane L holds row M=L&15; halves 0-7 = K {0..7}+(L<16?0:8)+k0,
//                  halves 8-15 = K {16..23}+(L<16?0:8)+k0
//   B (32x16 f16): lane L holds col N=L&15; halves h = K (L<16?0:8)+h+(h<8?0:8)+k0
//   C (16x16 f32): vgpr r, lane L -> row M=r+(L>=16?8:0), col N=L&15
// Weights pre-swizzled into B-fragment order: [(kstep*8+ntile)*32+lane]*16+h
// ---------------------------------------------------------------------------

typedef __attribute__((ext_vector_type(16))) _Float16 v16h;
typedef __attribute__((ext_vector_type(8)))  float    v8f;
typedef unsigned int u32t;
typedef __attribute__((ext_vector_type(4))) unsigned int v4u;
typedef __attribute__((ext_vector_type(4))) int          v4i;
typedef __attribute__((ext_vector_type(8))) int          v8i;

#define NN      50000
#define NE      600000
#define NG      128
#define ND_F    32
#define ED_F    16
#define H       128
#define T_OUT   128
#define LAYERS  4
#define NT      8              // 128 / 16 column tiles
#define WPB     4              // waves per block (block = 128 threads)
#define STG     (16 * H)       // per-wave staging tile (halves)

#define WAIT_DS() asm volatile("s_wait_dscnt 0" ::: "memory")

union AFrag { v16h h; float4 f4[2]; };

__device__ __forceinline__ v8f wmma16(const v16h a, const v16h b, v8f c) {
  return __builtin_amdgcn_wmma_f32_16x16x32_f16(false, a, false, b, (short)0, c, false, false);
}

__device__ __forceinline__ v16h load_bfrag(const _Float16* wf, int kstep, int ntile, int lane) {
  return *(const v16h*)(wf + ((((kstep * NT) + ntile) * 32 + lane) << 4));
}

__device__ __forceinline__ void pack8(v16h& h, int base, const float4 a, const float4 b) {
  h[base + 0] = (_Float16)a.x; h[base + 1] = (_Float16)a.y;
  h[base + 2] = (_Float16)a.z; h[base + 3] = (_Float16)a.w;
  h[base + 4] = (_Float16)b.x; h[base + 5] = (_Float16)b.y;
  h[base + 6] = (_Float16)b.z; h[base + 7] = (_Float16)b.w;
}

// ---------------------------------------------------------------------------
// TDM: 1-D contiguous global->LDS copy. D# per ISA 8.3/8.4:
//   group0: count=1 | lds_addr | global_addr[56:0] | type=2
//   group1: data_size=3 (8B units), tensor_dim0 = tile_dim0 = units, 1-D tile
__device__ __forceinline__ void tdm_load_1d(u32t lds_off, const void* gsrc, u32t bytes) {
  unsigned long long ga = (unsigned long long)(size_t)gsrc;
  u32t units = bytes >> 3;                         // 8-byte elements
  v4u g0;
  g0[0] = 1u;                                      // count=1 (valid), load, user
  g0[1] = lds_off;                                 // LDS byte address
  g0[2] = (u32t)ga;                                // global_addr[31:0]
  g0[3] = ((u32t)(ga >> 32) & 0x01FFFFFFu) | (2u << 30);  // ga[56:32] | type=2
  v8i g1;
  g1[0] = (int)(3u << 16);                         // wg_mask=0, data_size=3 (8B)
  g1[1] = (int)((units & 0xFFFFu) << 16);          // tensor_dim0[15:0]
  g1[2] = (int)(units >> 16);                      // tensor_dim0[31:16], tensor_dim1=0
  g1[3] = (int)((units & 0xFFFFu) << 16);          // tile_dim0 = units
  g1[4] = 0;                                       // tile_dim1=0 (1-D), tile_dim2=0
  g1[5] = (int)units;                              // tensor_dim0_stride (unused 1-D)
  g1[6] = 0;
  g1[7] = 0;
  v4i z4 = {0, 0, 0, 0};
#if defined(__clang_major__) && (__clang_major__ >= 23)
  v8i z8 = {0, 0, 0, 0, 0, 0, 0, 0};
  __builtin_amdgcn_tensor_load_to_lds(g0, g1, z4, z4, z8, 0);
#else
  __builtin_amdgcn_tensor_load_to_lds(g0, g1, z4, z4, 0);
#endif
}

__device__ __forceinline__ u32t lds_off_of(const void* p) {
  return (u32t)(size_t)p;   // LDS aperture: addr[31:0] is the LDS byte offset
}

// ---------------------------------------------------------------------------
// Weight pre-swizzle: row-major f32 [K x 128] -> f16 B-fragment order, K padded
__global__ void k_prep_wfrag(const float* __restrict__ W, int K, int Kpad,
                             _Float16* __restrict__ out) {
  int idx = blockIdx.x * blockDim.x + threadIdx.x;
  if (idx >= Kpad * H) return;
  int k = idx >> 7, col = idx & 127;
  float v = (k < K) ? W[k * H + col] : 0.0f;
  int kstep = k >> 5, kin = k & 31;
  int ntile = col >> 4;
  int lane  = (col & 15) + ((kin & 8) ? 16 : 0);
  int h     = (kin & 7) + ((kin & 16) ? 8 : 0);
  out[((((kstep * NT) + ntile) * 32 + lane) << 4) + h] = (_Float16)v;
}

// ---------------------------------------------------------------------------
__global__ void k_sq(const float* __restrict__ x, const int* __restrict__ src,
                     const int* __restrict__ dst, float* __restrict__ sq) {
  int e = blockIdx.x * blockDim.x + threadIdx.x;
  if (e >= NE) return;
  int s = src[e], d = dst[e];
  float dx = x[s * 3 + 0] - x[d * 3 + 0];
  float dy = x[s * 3 + 1] - x[d * 3 + 1];
  float dz = x[s * 3 + 2] - x[d * 3 + 2];
  sq[e] = dx * dx + dy * dy + dz * dz;
}

// ---------------------------------------------------------------------------
// f = relu(node_f @ W_in + b_in); emit f32 row-major + f16 A-fragment tiles
// LDS: [Win 4096h][stage WPB*2048h]  = 24 KB
__global__ void k_embed(const float* __restrict__ node_f, const float* __restrict__ b_in,
                        const _Float16* __restrict__ Win_f,
                        float* __restrict__ f32f, _Float16* __restrict__ fA) {
  extern __shared__ __align__(32) _Float16 smem[];
  _Float16* lW = smem;
  int wave = threadIdx.x >> 5, lane = threadIdx.x & 31;
  if (threadIdx.x < 32) {
    tdm_load_1d(lds_off_of(lW), Win_f, 4096 * 2);
    __builtin_amdgcn_s_wait_tensorcnt(0);
  }
  __syncthreads();

  int tile = blockIdx.x * WPB + wave;
  if (tile >= NN / 16) tile = NN / 16 - 1;   // tail clamp: duplicate plain stores
  _Float16* stage = smem + 4096 + wave * STG;
  int colr = lane & 15, hi = (lane >= 16), off = hi ? 8 : 0;

  const float* srow = node_f + (size_t)(tile * 16 + colr) * ND_F;
  AFrag a;
  pack8(a.h, 0, *(const float4*)(srow + off),      *(const float4*)(srow + off + 4));
  pack8(a.h, 8, *(const float4*)(srow + 16 + off), *(const float4*)(srow + 16 + off + 4));

  float bc[NT];
#pragma unroll
  for (int n = 0; n < NT; n++) bc[n] = b_in[n * 16 + colr];

  v8f acc[NT] = {};
#pragma unroll
  for (int n = 0; n < NT; n++)
    acc[n] = wmma16(a.h, load_bfrag(lW, 0, n, lane), acc[n]);

#pragma unroll
  for (int n = 0; n < NT; n++)
#pragma unroll
    for (int r = 0; r < 8; r++) {
      int M = r + hi * 8, col = n * 16 + colr;
      float v = fmaxf(acc[n][r] + bc[n], 0.0f);
      f32f[(size_t)(tile * 16 + M) * H + col] = v;
      stage[M * H + col] = (_Float16)v;
    }
  WAIT_DS();
#pragma unroll
  for (int k = 0; k < 4; k++) {
    const _Float16* sp = stage + colr * H + k * 32 + off;
    float4* dp = (float4*)fA + (((tile * 4 + k) * 32 + lane) << 1);
    dp[0] = *(const float4*)sp;
    dp[1] = *(const float4*)(sp + 16);
  }
}

// ---------------------------------------------------------------------------
// P = f@W1a, Q = f@W1b  (dense node GEMMs). LDS: [Wa 16384h][Wb 16384h] = 64 KB
__global__ void k_node_pq(const _Float16* __restrict__ fA,
                          const _Float16* __restrict__ Wa_f, const _Float16* __restrict__ Wb_f,
                          float* __restrict__ P, float* __restrict__ Q) {
  extern __shared__ __align__(32) _Float16 smem[];
  _Float16* lWa = smem;
  _Float16* lWb = smem + 16384;
  int wave = threadIdx.x >> 5, lane = threadIdx.x & 31;
  if (threadIdx.x < 32) {
    tdm_load_1d(lds_off_of(lWa), Wa_f, 16384 * 2);
    tdm_load_1d(lds_off_of(lWb), Wb_f, 16384 * 2);
    __builtin_amdgcn_s_wait_tensorcnt(0);
  }
  __syncthreads();

  int tile = blockIdx.x * WPB + wave;
  if (tile >= NN / 16) tile = NN / 16 - 1;
  int colr = lane & 15, hi = (lane >= 16);

  v8f ap[NT] = {}, aq[NT] = {};
#pragma unroll
  for (int k = 0; k < 4; k++) {
    v16h a = *(const v16h*)(fA + ((((tile * 4 + k) * 32) + lane) << 4));
#pragma unroll
    for (int n = 0; n < NT; n++) {
      ap[n] = wmma16(a, load_bfrag(lWa, k, n, lane), ap[n]);
      aq[n] = wmma16(a, load_bfrag(lWb, k, n, lane), aq[n]);
    }
  }
#pragma unroll
  for (int n = 0; n < NT; n++)
#pragma unroll
    for (int r = 0; r < 8; r++) {
      size_t row = (size_t)(tile * 16 + r + hi * 8);
      int col = n * 16 + colr;
      P[row * H + col] = ap[n][r];
      Q[row * H + col] = aq[n][r];
    }
}

// ---------------------------------------------------------------------------
// Fused edge kernel. LDS: [We 4096h][W1c 16384h][W2 16384h][stage 8192h] = 88 KB
//   w  = relu(edge_w @ W_e + b_e)
//   h1 = relu(w @ W1c + P[src] + Q[dst] + sq*w1d + b1)
//   m  = relu(h1 @ W2 + b2)  -> atomic scatter-add into msum[dst]
__global__ void k_msg(const float* __restrict__ edge_w,
                      const _Float16* __restrict__ We_f, const float* __restrict__ b_e,
                      const _Float16* __restrict__ W1c_f, const float* __restrict__ w1d,
                      const float* __restrict__ b1,
                      const _Float16* __restrict__ W2_f, const float* __restrict__ b2,
                      const float* __restrict__ P, const float* __restrict__ Q,
                      const int* __restrict__ src, const int* __restrict__ dst,
                      const float* __restrict__ sq, float* __restrict__ msum) {
  extern __shared__ __align__(32) _Float16 smem[];
  _Float16* lWe  = smem;
  _Float16* lW1c = smem + 4096;
  _Float16* lW2  = smem + 4096 + 16384;
  int wave = threadIdx.x >> 5, lane = threadIdx.x & 31;
  if (threadIdx.x < 32) {
    tdm_load_1d(lds_off_of(lWe),  We_f,  4096 * 2);
    tdm_load_1d(lds_off_of(lW1c), W1c_f, 16384 * 2);
    tdm_load_1d(lds_off_of(lW2),  W2_f,  16384 * 2);
    __builtin_amdgcn_s_wait_tensorcnt(0);
  }
  __syncthreads();

  int tile = blockIdx.x * WPB + wave;
  if (tile >= NE / 16) return;    // grid divides exactly; guard after barrier
  _Float16* stage = smem + 4096 + 2 * 16384 + wave * STG;
  int e0 = tile * 16;
  int colr = lane & 15, hi = (lane >= 16), off = hi ? 8 : 0;

  int se[8], de[8]; float sqe[8];
#pragma unroll
  for (int r = 0; r < 8; r++) {
    int e = e0 + r + hi * 8;
    se[r] = src[e]; de[r] = dst[e]; sqe[r] = sq[e];
  }
#pragma unroll
  for (int r = 0; r < 8; r++) {               // warm caches for the gathers
    __builtin_prefetch(P + (size_t)se[r] * H, 0, 0);
    __builtin_prefetch(Q + (size_t)de[r] * H, 0, 0);
  }

  float bec[NT], b1c[NT], b2c[NT], w1dc[NT];
#pragma unroll
  for (int n = 0; n < NT; n++) {
    int c = n * 16 + colr;
    bec[n] = b_e[c]; b1c[n] = b1[c]; b2c[n] = b2[c]; w1dc[n] = w1d[c];
  }

  // ---- stage 0: w = relu(edge_w @ W_e + b_e), ED=16 padded to K=32 ----
  const float* ew = edge_w + (size_t)(e0 + colr) * ED_F;
  AFrag a;
  pack8(a.h, 0, *(const float4*)(ew + off), *(const float4*)(ew + off + 4));
#pragma unroll
  for (int i = 8; i < 16; i++) a.h[i] = (_Float16)0.0f;

  v8f acc[NT] = {};
#pragma unroll
  for (int n = 0; n < NT; n++)
    acc[n] = wmma16(a.h, load_bfrag(lWe, 0, n, lane), acc[n]);
#pragma unroll
  for (int n = 0; n < NT; n++)
#pragma unroll
    for (int r = 0; r < 8; r++)
      stage[(r + hi * 8) * H + n * 16 + colr] =
          (_Float16)fmaxf(acc[n][r] + bec[n], 0.0f);
  WAIT_DS();

  // ---- stage 1: h1 = relu(w @ W1c + P[src] + Q[dst] + sq*w1d + b1) ----
#pragma unroll
  for (int n = 0; n < NT; n++) acc[n] = (v8f){};
#pragma unroll
  for (int k = 0; k < 4; k++) {
    AFrag aw;
    const _Float16* sp = stage + colr * H + k * 32 + off;
    aw.f4[0] = *(const float4*)sp;
    aw.f4[1] = *(const float4*)(sp + 16);
#pragma unroll
    for (int n = 0; n < NT; n++)
      acc[n] = wmma16(aw.h, load_bfrag(lW1c, k, n, lane), acc[n]);
  }
  WAIT_DS();
#pragma unroll
  for (int r = 0; r < 8; r++) {
    const float* pr = P + (size_t)se[r] * H;
    const float* qr = Q + (size_t)de[r] * H;
#pragma unroll
    for (int n = 0; n < NT; n++) {
      int c = n * 16 + colr;
      float v = acc[n][r] + pr[c] + qr[c] + sqe[r] * w1dc[n] + b1c[n];
      stage[(r + hi * 8) * H + c] = (_Float16)fmaxf(v, 0.0f);
    }
  }
  WAIT_DS();

  // ---- stage 2: m = relu(h1 @ W2 + b2) ----
#pragma unroll
  for (int n = 0; n < NT; n++) acc[n] = (v8f){};
#pragma unroll
  for (int k = 0; k < 4; k++) {
    AFrag ah;
    const _Float16* sp = stage + colr * H + k * 32 + off;
    ah.f4[0] = *(const float4*)sp;
    ah.f4[1] = *(const float4*)(sp + 16);
#pragma unroll
    for (int n = 0; n < NT; n++)
      acc[n] = wmma16(ah.h, load_bfrag(lW2, k, n, lane), acc[n]);
  }

#pragma unroll
  for (int r = 0; r < 8; r++) {
    float* dr = msum + (size_t)de[r] * H;
#pragma unroll
    for (int n = 0; n < NT; n++) {
      float v = fmaxf(acc[n][r] + b2c[n], 0.0f);
      atomicAdd(dr + n * 16 + colr, v);
    }
  }
}

// ---------------------------------------------------------------------------
// u = msum + f ; f_new = relu(relu(u@U1+ub1)@U2+ub2) ; emit f32 + f16 A-tiles
// LDS: [U1 16384h][U2 16384h][stage 8192h] = 80 KB
__global__ void k_update(const float* __restrict__ msum, float* __restrict__ f32f,
                         const _Float16* __restrict__ U1_f, const float* __restrict__ ub1,
                         const _Float16* __restrict__ U2_f, const float* __restrict__ ub2,
                         _Float16* __restrict__ fA) {
  extern __shared__ __align__(32) _Float16 smem[];
  _Float16* lU1 = smem;
  _Float16* lU2 = smem + 16384;
  int wave = threadIdx.x >> 5, lane = threadIdx.x & 31;
  if (threadIdx.x < 32) {
    tdm_load_1d(lds_off_of(lU1), U1_f, 16384 * 2);
    tdm_load_1d(lds_off_of(lU2), U2_f, 16384 * 2);
    __builtin_amdgcn_s_wait_tensorcnt(0);
  }
  __syncthreads();

  int tile = blockIdx.x * WPB + wave;
  if (tile >= NN / 16) tile = NN / 16 - 1;
  _Float16* stage = smem + 2 * 16384 + wave * STG;
  int colr = lane & 15, hi = (lane >= 16), off = hi ? 8 : 0;

  float b1c[NT], b2c[NT];
#pragma unroll
  for (int n = 0; n < NT; n++) {
    b1c[n] = ub1[n * 16 + colr];
    b2c[n] = ub2[n * 16 + colr];
  }

  const float* mrow = msum + (size_t)(tile * 16 + colr) * H;
  const float* frow = f32f + (size_t)(tile * 16 + colr) * H;

  v8f acc[NT] = {};
#pragma unroll
  for (int k = 0; k < 4; k++) {
    AFrag a;
#pragma unroll
    for (int c = 0; c < 2; c++) {
      int base = k * 32 + c * 16 + off;
      float4 m0 = *(const float4*)(mrow + base);
      float4 m1 = *(const float4*)(mrow + base + 4);
      float4 f0 = *(const float4*)(frow + base);
      float4 f1 = *(const float4*)(frow + base + 4);
      float4 s0 = {m0.x + f0.x, m0.y + f0.y, m0.z + f0.z, m0.w + f0.w};
      float4 s1 = {m1.x + f1.x, m1.y + f1.y, m1.z + f1.z, m1.w + f1.w};
      pack8(a.h, c * 8, s0, s1);
    }
#pragma unroll
    for (int n = 0; n < NT; n++)
      acc[n] = wmma16(a.h, load_bfrag(lU1, k, n, lane), acc[n]);
  }
#pragma unroll
  for (int n = 0; n < NT; n++)
#pragma unroll
    for (int r = 0; r < 8; r++)
      stage[(r + hi * 8) * H + n * 16 + colr] =
          (_Float16)fmaxf(acc[n][r] + b1c[n], 0.0f);
  WAIT_DS();

#pragma unroll
  for (int n = 0; n < NT; n++) acc[n] = (v8f){};
#pragma unroll
  for (int k = 0; k < 4; k++) {
    AFrag ah;
    const _Float16* sp = stage + colr * H + k * 32 + off;
    ah.f4[0] = *(const float4*)sp;
    ah.f4[1] = *(const float4*)(sp + 16);
#pragma unroll
    for (int n = 0; n < NT; n++)
      acc[n] = wmma16(ah.h, load_bfrag(lU2, k, n, lane), acc[n]);
  }
  WAIT_DS();
#pragma unroll
  for (int n = 0; n < NT; n++)
#pragma unroll
    for (int r = 0; r < 8; r++) {
      int M = r + hi * 8, col = n * 16 + colr;
      float v = fmaxf(acc[n][r] + b2c[n], 0.0f);
      f32f[(size_t)(tile * 16 + M) * H + col] = v;  // in-place: wave owns rows
      stage[M * H + col] = (_Float16)v;
    }
  WAIT_DS();
#pragma unroll
  for (int k = 0; k < 4; k++) {
    const _Float16* sp = stage + colr * H + k * 32 + off;
    float4* dp = (float4*)fA + (((tile * 4 + k) * 32 + lane) << 1);
    dp[0] = *(const float4*)sp;
    dp[1] = *(const float4*)(sp + 16);
  }
}

// ---------------------------------------------------------------------------
__global__ void k_readout_acc(const float* __restrict__ f, const int* __restrict__ gid,
                              float* __restrict__ gsum, unsigned int* __restrict__ gmax,
                              float* __restrict__ cnt) {
  int idx = blockIdx.x * blockDim.x + threadIdx.x;
  if (idx >= NN * H) return;
  int node = idx >> 7, col = idx & 127;
  int g = gid[node];
  float v = f[idx];
  atomicAdd(&gsum[g * H + col], v);
  atomicMax(&gmax[g * H + col], __float_as_uint(v));   // v >= 0 after relu
  if (col == 0) atomicAdd(&cnt[g], 1.0f);
}

__global__ void k_readout_out(const float* __restrict__ gsum, const unsigned int* __restrict__ gmax,
                              const float* __restrict__ cnt, const float* __restrict__ W_out,
                              const float* __restrict__ b_out, float* __restrict__ out) {
  int g = blockIdx.x, t = threadIdx.x;
  float inv = 1.0f / fmaxf(cnt[g], 1.0f);
  float acc = b_out[t];
  for (int c = 0; c < H; c++) acc += gsum[g * H + c] * inv * W_out[c * T_OUT + t];
  for (int c = 0; c < H; c++) acc += __uint_as_float(gmax[g * H + c]) * W_out[(H + c) * T_OUT + t];
  out[g * T_OUT + t] = acc;
}

// ---------------------------------------------------------------------------
extern "C" void kernel_launch(void* const* d_in, const int* in_sizes, int n_in,
                              void* d_out, int out_size, void* d_ws, size_t ws_size,
                              hipStream_t stream) {
  const float* node_f = (const float*)d_in[0];
  const float* node_x = (const float*)d_in[1];
  const float* edge_w = (const float*)d_in[2];
  const int*   src    = (const int*)d_in[3];
  const int*   dst    = (const int*)d_in[4];
  const int*   gid    = (const int*)d_in[5];
  const float* W_in   = (const float*)d_in[6];
  const float* b_in   = (const float*)d_in[7];
  const float* W_e    = (const float*)d_in[8];
  const float* b_e    = (const float*)d_in[9];
  const float* msg_W1 = (const float*)d_in[10];
  const float* msg_b1 = (const float*)d_in[11];
  const float* msg_W2 = (const float*)d_in[12];
  const float* msg_b2 = (const float*)d_in[13];
  const float* upd_W1 = (const float*)d_in[14];
  const float* upd_b1 = (const float*)d_in[15];
  const float* upd_W2 = (const float*)d_in[16];
  const float* upd_b2 = (const float*)d_in[17];
  const float* W_out  = (const float*)d_in[18];
  const float* b_out  = (const float*)d_in[19];

  char* ws = (char*)d_ws;
  size_t o = 0;
  auto carve = [&](size_t bytes) -> char* {
    char* p = ws + o;
    o += (bytes + 255) & ~(size_t)255;
    return p;
  };
  float*        f32f  = (float*)carve((size_t)NN * H * 4);
  float*        msum  = (float*)carve((size_t)NN * H * 4);
  float*        P     = (float*)carve((size_t)NN * H * 4);
  float*        Q     = (float*)carve((size_t)NN * H * 4);
  _Float16*     fA    = (_Float16*)carve((size_t)NN * H * 2);
  float*        sq    = (float*)carve((size_t)NE * 4);
  float*        gsum  = (float*)carve((size_t)NG * H * 4);
  unsigned int* gmax  = (unsigned int*)carve((size_t)NG * H * 4);
  float*        cnt   = (float*)carve((size_t)NG * 4);
  _Float16*     Win_f = (_Float16*)carve((size_t)32 * H * 2);
  _Float16*     We_f  = (_Float16*)carve((size_t)32 * H * 2);
  _Float16*     wf    = (_Float16*)carve((size_t)LAYERS * 6 * H * H * 2);

  const int PW = 256;
  k_prep_wfrag<<<(32 * H + PW - 1) / PW, PW, 0, stream>>>(W_in, 32, 32, Win_f);
  k_prep_wfrag<<<(32 * H + PW - 1) / PW, PW, 0, stream>>>(W_e, 16, 32, We_f);
  for (int i = 0; i < LAYERS; i++) {
    const float* m1 = msg_W1 + (size_t)i * 385 * H;
    _Float16* base = wf + (size_t)i * 6 * H * H;
    int nb = (H * H + PW - 1) / PW;
    k_prep_wfrag<<<nb, PW, 0, stream>>>(m1 + 0 * H * H, H, H, base + 0 * H * H); // W1a
    k_prep_wfrag<<<nb, PW, 0, stream>>>(m1 + 1 * H * H, H, H, base + 1 * H * H); // W1b
    k_prep_wfrag<<<nb, PW, 0, stream>>>(m1 + 2 * H * H, H, H, base + 2 * H * H); // W1c
    k_prep_wfrag<<<nb, PW, 0, stream>>>(msg_W2 + (size_t)i * H * H, H, H, base + 3 * H * H);
    k_prep_wfrag<<<nb, PW, 0, stream>>>(upd_W1 + (size_t)i * H * H, H, H, base + 4 * H * H);
    k_prep_wfrag<<<nb, PW, 0, stream>>>(upd_W2 + (size_t)i * H * H, H, H, base + 5 * H * H);
  }

  k_sq<<<(NE + PW - 1) / PW, PW, 0, stream>>>(node_x, src, dst, sq);

  const int NODE_TILES = NN / 16, EDGE_TILES = NE / 16;
  const int NODE_BLKS = (NODE_TILES + WPB - 1) / WPB;
  const int EDGE_BLKS = (EDGE_TILES + WPB - 1) / WPB;

  const size_t SM_EMBED = (size_t)(4096 + WPB * STG) * 2;            // 24 KB
  const size_t SM_PQ    = (size_t)(2 * 16384) * 2;                   // 64 KB
  const size_t SM_MSG   = (size_t)(4096 + 2 * 16384 + WPB * STG) * 2;// 88 KB
  const size_t SM_UPD   = (size_t)(2 * 16384 + WPB * STG) * 2;       // 80 KB

  k_embed<<<NODE_BLKS, 32 * WPB, SM_EMBED, stream>>>(node_f, b_in, Win_f, f32f, fA);

  for (int i = 0; i < LAYERS; i++) {
    _Float16* base = wf + (size_t)i * 6 * H * H;
    const float* m1 = msg_W1 + (size_t)i * 385 * H;
    k_node_pq<<<NODE_BLKS, 32 * WPB, SM_PQ, stream>>>(
        fA, base + 0 * H * H, base + 1 * H * H, P, Q);
    hipMemsetAsync(msum, 0, (size_t)NN * H * 4, stream);
    k_msg<<<EDGE_BLKS, 32 * WPB, SM_MSG, stream>>>(
        edge_w, We_f, b_e,
        base + 2 * H * H, m1 + (size_t)384 * H, msg_b1 + (size_t)i * H,
        base + 3 * H * H, msg_b2 + (size_t)i * H,
        P, Q, src, dst, sq, msum);
    k_update<<<NODE_BLKS, 32 * WPB, SM_UPD, stream>>>(
        msum, f32f, base + 4 * H * H, upd_b1 + (size_t)i * H,
        base + 5 * H * H, upd_b2 + (size_t)i * H, fA);
  }

  hipMemsetAsync(gsum, 0, (size_t)NG * H * 4, stream);
  hipMemsetAsync(gmax, 0, (size_t)NG * H * 4, stream);
  hipMemsetAsync(cnt, 0, (size_t)NG * 4, stream);
  k_readout_acc<<<(NN * H + PW - 1) / PW, PW, 0, stream>>>(f32f, gid, gsum, gmax, cnt);
  k_readout_out<<<NG, T_OUT, 0, stream>>>(gsum, gmax, cnt, W_out, b_out, (float*)d_out);
}
